// Network_77833397338501
// MI455X (gfx1250) — matrix operand order repaired and checked by reading, not compiled, not run on hardware
//
#include <hip/hip_runtime.h>

typedef __attribute__((ext_vector_type(16))) _Float16 v16h;
typedef __attribute__((ext_vector_type(8)))  float    v8f;
typedef __attribute__((ext_vector_type(4)))  unsigned int u32x4;
typedef __attribute__((ext_vector_type(8)))  int      i32x8;
typedef __attribute__((ext_vector_type(4)))  int      i32x4;

#define EPSV    1e-5f
#define WTHRESH 0.05f
#define USE_TDM 1

// ---------------------------------------------------------------------------
// Kernels
// ---------------------------------------------------------------------------

// Stem: 3x3 conv, 3 -> 48 channels, stride 1, pad 1 (direct, tiny K)
__global__ void stem_conv_k(const float* __restrict__ X, const float* __restrict__ W,
                            float* __restrict__ Y, int N, int H, int Wd) {
  long idx = (long)blockIdx.x * blockDim.x + threadIdx.x;
  long total = (long)N * 48 * H * Wd;
  if (idx >= total) return;
  int w  = (int)(idx % Wd);
  int h  = (int)((idx / Wd) % H);
  int co = (int)((idx / ((long)Wd * H)) % 48);
  int n  = (int)(idx / ((long)Wd * H * 48));
  float acc = 0.f;
  #pragma unroll
  for (int ci = 0; ci < 3; ++ci)
    #pragma unroll
    for (int kh = 0; kh < 3; ++kh)
      #pragma unroll
      for (int kw = 0; kw < 3; ++kw) {
        int hi = h + kh - 1, wi = w + kw - 1;
        if (hi >= 0 && hi < H && wi >= 0 && wi < Wd)
          acc += X[((long)(n * 3 + ci) * H + hi) * Wd + wi] *
                 W[((co * 3 + ci) * 3 + kh) * 3 + kw];
      }
  Y[idx] = acc;
}

// BatchNorm stats: one block per channel, mean + biased var over (N,H,W)
__global__ void bn_stats_k(const float* __restrict__ X, float* __restrict__ mv,
                           int C, int N, int S) {
  int c = blockIdx.x;
  __shared__ float ss[256], sq[256];
  float s = 0.f, q = 0.f;
  for (int i = threadIdx.x; i < N * S; i += blockDim.x) {
    int n = i / S, p = i - n * S;
    float v = X[((long)n * C + c) * S + p];
    s += v; q += v * v;
  }
  ss[threadIdx.x] = s; sq[threadIdx.x] = q;
  __syncthreads();
  for (int off = 128; off > 0; off >>= 1) {
    if ((int)threadIdx.x < off) {
      ss[threadIdx.x] += ss[threadIdx.x + off];
      sq[threadIdx.x] += sq[threadIdx.x + off];
    }
    __syncthreads();
  }
  if (threadIdx.x == 0) {
    float inv = 1.f / (float)(N * S);
    float m = ss[0] * inv;
    mv[2 * c]     = m;
    mv[2 * c + 1] = sq[0] * inv - m * m;
  }
}

// BatchNorm apply in place; optional affine (g,b may be null)
__global__ void bn_apply_k(float* __restrict__ X, const float* __restrict__ mv,
                           const float* __restrict__ g, const float* __restrict__ b,
                           int C, int N, int S) {
  long idx = (long)blockIdx.x * blockDim.x + threadIdx.x;
  long total = (long)N * C * S;
  if (idx >= total) return;
  int c = (int)((idx / S) % C);
  float y = (X[idx] - mv[2 * c]) * rsqrtf(mv[2 * c + 1] + EPSV);
  if (g) y = y * g[c] + b[c];
  X[idx] = y;
}

// Pointwise (1x1) convolution as WMMA GEMM:
//   Y[coBase+co, pixel] = sum_ci W[co,ci] * act(X[ciBase+ci, pixel_in])
// MT = Cout/16 (Cout in {16,32,64}). Whole W matrix (Cout x Cin fp32) is staged
// into LDS once per block — via the Tensor Data Mover when available — then
// each wave computes MT 16x16 output tiles against one shared B fragment per
// K-step, with MT v_wmma_f32_16x16x32_f16 issues per iteration.
// Requires: (N*Ho*Wo) % 64 == 0, Cin <= 192. Block = 128 threads (4 waves).
template <int MT>
__global__ void pw_conv_wmma_k(const float* __restrict__ X, const float* __restrict__ W,
                               float* __restrict__ Y,
                               int Cin, int CinTot, int ciBase,
                               int CoutTot, int coBase,
                               int N, int H, int Wd, int stride, int off,
                               int Ho, int Wo, int relu) {
  __shared__ float sW[MT * 16 * 192];
  const int Cout = MT * 16;

  // ---- stage W (Cout x Cin fp32, row-major, dense) into LDS ----
#if USE_TDM && __has_builtin(__builtin_amdgcn_tensor_load_to_lds)
  if (threadIdx.x < 32) {  // one wave issues the DMA (TDM ignores EXEC)
    unsigned long long ga = (unsigned long long)(const void*)W;
    u32x4 g0;
    g0[0] = 1u;                                        // count=1 (valid user D#)
    g0[1] = 0u;                                        // lds_addr: sW at LDS offset 0
    g0[2] = (unsigned)(ga & 0xffffffffull);            // global_addr[31:0]
    g0[3] = (unsigned)((ga >> 32) & 0x01ffffffull)     // global_addr[56:32]
            | (2u << 30);                              // type=2 ("image")
    i32x8 g1;
    g1[0] = (int)(2u << 16);                           // data_size=2 -> 4-byte elems
    g1[1] = (int)(((unsigned)Cin & 0xffffu) << 16);    // tensor_dim0[15:0]
    g1[2] = (int)((((unsigned)Cin >> 16) & 0xffffu)    // tensor_dim0[31:16]
            | (((unsigned)Cout & 0xffffu) << 16));     // tensor_dim1[15:0]
    g1[3] = (int)((((unsigned)Cout >> 16) & 0xffffu)   // tensor_dim1[31:16]
            | (((unsigned)Cin & 0xffffu) << 16));      // tile_dim0 = Cin
    g1[4] = (int)((unsigned)Cout & 0xffffu);           // tile_dim1 = Cout, tile_dim2=0
    g1[5] = Cin;                                       // tensor_dim0_stride[31:0]
    g1[6] = 0;                                         // stride hi / dim1_stride lo
    g1[7] = 0;                                         // dim1_stride hi
    i32x4 gz4 = {0, 0, 0, 0};                          // 2D tensor: groups 2/3 unused
    i32x8 gz8 = {0, 0, 0, 0, 0, 0, 0, 0};
    __builtin_amdgcn_tensor_load_to_lds(g0, g1, gz4, gz4, gz8, 0);
    __builtin_amdgcn_s_wait_tensorcnt(0);
  }
  __syncthreads();
#else
  for (int i = threadIdx.x; i < Cout * Cin; i += blockDim.x) sW[i] = W[i];
  __syncthreads();
#endif

  int lane = threadIdx.x & 31;
  int wv   = threadIdx.x >> 5;
  int pt   = (blockIdx.x * 4 + wv) * 16;   // pixel tile base
  int m    = lane & 15;
  int kg   = lane >> 4;                    // K-half selector

  int HW = H * Wd;
  int oS = Ho * Wo;
  int p   = pt + m;                        // this lane's pixel (B column / D column)
  int n   = p / oS;
  int rem = p - n * oS;
  int ho  = rem / Wo;
  int wo  = rem - ho * Wo;
  int hi  = ho * stride + off;
  int wi  = wo * stride + off;
  long xbase = ((long)n * CinTot + ciBase) * HW + (long)hi * Wd + wi;

  v8f zero = {};
  v8f acc[MT];
  #pragma unroll
  for (int q = 0; q < MT; ++q) acc[q] = zero;

  for (int k0 = 0; k0 < Cin; k0 += 32) {
    // B fragment (32x16 f16): lane-half 0 holds K 0..7,16..23; half 1 holds K 8..15,24..31
    v16h b;
    #pragma unroll
    for (int e = 0; e < 16; ++e) {
      int kb = k0 + ((e < 8) ? e : e + 8) + kg * 8;
      float xv = (kb < Cin) ? X[xbase + (long)kb * HW] : 0.f;
      if (relu) xv = fmaxf(xv, 0.f);
      b[e] = (_Float16)xv;
    }
    if (k0 + 32 < Cin)
      __builtin_prefetch(X + xbase + (long)(k0 + 32) * HW, 0, 0);
    #pragma unroll
    for (int q = 0; q < MT; ++q) {
      // A fragment (16x32 f16) for Cout tile q, read from LDS
      v16h a;
      #pragma unroll
      for (int e = 0; e < 16; ++e) {
        int v  = e >> 1, lh = e & 1;
        int ka = ((v < 4) ? (v * 2 + lh) : (16 + (v - 4) * 2 + lh)) + kg * 8;
        int k  = k0 + ka;
        float wvv = (k < Cin) ? sW[(q * 16 + m) * Cin + k] : 0.f;
        a[e] = (_Float16)wvv;
      }
      acc[q] = __builtin_amdgcn_wmma_f32_16x16x32_f16(false, a, false, b,
                                                      (short)0, acc[q], false, false);
    }
  }

  // D fragment: VGPR r -> row r + 8*kg, column = lane&15 (this lane's pixel)
  #pragma unroll
  for (int q = 0; q < MT; ++q)
    #pragma unroll
    for (int r = 0; r < 8; ++r) {
      int row = q * 16 + r + 8 * kg;
      Y[((long)n * CoutTot + coBase + row) * oS + rem] = acc[q][r];
    }
}

// Depthwise KxK conv (groups == C), optional ReLU on input
__global__ void dw_conv_k(const float* __restrict__ X, const float* __restrict__ W,
                          float* __restrict__ Y,
                          int C, int CinTot, int ciBase,
                          int N, int H, int Wd, int Ho, int Wo,
                          int k, int stride, int pad, int dil, int relu) {
  long idx = (long)blockIdx.x * blockDim.x + threadIdx.x;
  long total = (long)N * C * Ho * Wo;
  if (idx >= total) return;
  int wo = (int)(idx % Wo);
  int ho = (int)((idx / Wo) % Ho);
  int c  = (int)((idx / ((long)Wo * Ho)) % C);
  int n  = (int)(idx / ((long)Wo * Ho * C));
  long xb = ((long)n * CinTot + ciBase + c) * H * Wd;
  float acc = 0.f;
  for (int kh = 0; kh < k; ++kh)
    for (int kw = 0; kw < k; ++kw) {
      int hi = ho * stride - pad + kh * dil;
      int wi = wo * stride - pad + kw * dil;
      if (hi >= 0 && hi < H && wi >= 0 && wi < Wd) {
        float xv = X[xb + (long)hi * Wd + wi];
        if (relu) xv = fmaxf(xv, 0.f);
        acc += xv * W[(c * k + kh) * k + kw];
      }
    }
  Y[idx] = acc;
}

// 3x3 pool, pad 1. isMax=1 -> max pool; isMax=0 -> avg (count_include_pad=False)
__global__ void pool3_k(const float* __restrict__ X, float* __restrict__ Y,
                        int C, int CinTot, int ciBase,
                        int N, int H, int Wd, int Ho, int Wo, int stride, int isMax) {
  long idx = (long)blockIdx.x * blockDim.x + threadIdx.x;
  long total = (long)N * C * Ho * Wo;
  if (idx >= total) return;
  int wo = (int)(idx % Wo);
  int ho = (int)((idx / Wo) % Ho);
  int c  = (int)((idx / ((long)Wo * Ho)) % C);
  int n  = (int)(idx / ((long)Wo * Ho * C));
  long xb = ((long)n * CinTot + ciBase + c) * H * Wd;
  float acc = isMax ? -INFINITY : 0.f;
  int cnt = 0;
  #pragma unroll
  for (int kh = 0; kh < 3; ++kh)
    #pragma unroll
    for (int kw = 0; kw < 3; ++kw) {
      int hi = ho * stride - 1 + kh;
      int wi = wo * stride - 1 + kw;
      if (hi >= 0 && hi < H && wi >= 0 && wi < Wd) {
        float v = X[xb + (long)hi * Wd + wi];
        if (isMax) acc = fmaxf(acc, v); else { acc += v; ++cnt; }
      }
    }
  Y[idx] = isMax ? acc : acc / (float)cnt;
}

// OUT[slice] += w_eff * T[slice], w_eff = (wA[prim]+wB[prim] < 0.05) ? 0 : sum
__global__ void axpy_mixed_k(float* __restrict__ OUT, const float* __restrict__ T,
                             const float* __restrict__ wA, const float* __restrict__ wB,
                             int prim, int C, int N, int S,
                             int outCTot, int outCBase, int inCTot, int inCBase) {
  long idx = (long)blockIdx.x * blockDim.x + threadIdx.x;
  long total = (long)N * C * S;
  if (idx >= total) return;
  float s = wA[prim] + wB[prim];
  if (s < WTHRESH) s = 0.f;
  int p = (int)(idx % S);
  int c = (int)((idx / S) % C);
  int n = (int)(idx / ((long)S * C));
  OUT[((long)n * outCTot + outCBase + c) * S + p] +=
      s * T[((long)n * inCTot + inCBase + c) * S + p];
}

// Global average over spatial + FC head: out[n,cls]
__global__ void head_fc_k(const float* __restrict__ X, const float* __restrict__ FW,
                          const float* __restrict__ FB, float* __restrict__ OUT,
                          int N, int C, int S) {
  int idx = blockIdx.x * blockDim.x + threadIdx.x;
  if (idx >= N * 10) return;
  int n = idx / 10, cls = idx % 10;
  float acc = FB[cls];
  float inv = 1.f / (float)S;
  for (int c = 0; c < C; ++c) {
    const float* xp = X + ((long)n * C + c) * S;
    float m = 0.f;
    for (int p = 0; p < S; ++p) m += xp[p];
    acc += m * inv * FW[cls * C + c];
  }
  OUT[idx] = acc;
}

// ---------------------------------------------------------------------------
// Host orchestration
// ---------------------------------------------------------------------------

struct St { const float* p; int Ctot; int cBase; int H; };

struct OpParams {
  const float *s3dw1,*s3pw1,*s3dw2,*s3pw2;
  const float *d3dw,*d3pw;
  const float *s5dw1,*s5pw1,*s5dw2,*s5pw2;
  const float *d5dw,*d5pw;
  const float *skc1,*skc2;
};

extern "C" void kernel_launch(void* const* d_in, const int* in_sizes, int n_in,
                              void* d_out, int out_size, void* d_ws, size_t ws_size,
                              hipStream_t stream) {
  (void)in_sizes; (void)n_in; (void)out_size; (void)ws_size;
  const int N = 64;

  // ---- workspace arena (floats) ----
  float* ws = (float*)d_ws;
  size_t o = 0;
  auto alloc = [&](size_t nf) { float* r = ws + o; o += nf; return r; };
  const size_t PER = 64ull * 48 * 32 * 32;           // 3,145,728 floats
  float* B0  = alloc(PER);
  float* B1  = alloc(PER);
  float* B2  = alloc(PER);
  float* s0p = alloc(64ull * 32 * 32 * 32);          // preprocessed s0
  float* s1p = alloc(64ull * 32 * 32 * 32);          // preprocessed s1
  float* t   = alloc(64ull * 16 * 32 * 32);          // op result
  float* u1  = alloc(64ull * 16 * 32 * 32);          // sep/dil intermediates
  float* u2  = alloc(64ull * 16 * 32 * 32);
  float* mv  = alloc(512);                           // bn mean/var

  // ---- input leaf walk (dict insertion order, depth-first) ----
  int leaf = 0;
  auto nxt = [&]() { return (const float*)d_in[leaf++]; };
  const float* data   = nxt();
  const float* aw     = nxt();
  const float* stem_w = nxt();
  const float* stem_g = nxt();
  const float* stem_b = nxt();

  auto NB = [](long nthr) { return (unsigned)((nthr + 255) / 256); };

  auto bnl = [&](float* X, int C, int S, const float* g = nullptr, const float* b = nullptr) {
    bn_stats_k<<<C, 256, 0, stream>>>(X, mv, C, N, S);
    long tot = (long)N * C * S;
    bn_apply_k<<<NB(tot), 256, 0, stream>>>(X, mv, g, b, C, N, S);
  };

  auto pwl = [&](const float* X, int Cin, int CinTot, int ciBase, int H,
                 const float* Wp, int Cout, int CoutTot, int coBase,
                 float* Y, int stride, int off, int relu) {
    int Ho = H / stride;
    int P  = N * Ho * Ho;
    dim3 g2((unsigned)(P / 64), 1, 1);
    if (Cout == 16)
      pw_conv_wmma_k<1><<<g2, 128, 0, stream>>>(X, Wp, Y, Cin, CinTot, ciBase,
                                                CoutTot, coBase, N, H, H, stride, off,
                                                Ho, Ho, relu);
    else if (Cout == 32)
      pw_conv_wmma_k<2><<<g2, 128, 0, stream>>>(X, Wp, Y, Cin, CinTot, ciBase,
                                                CoutTot, coBase, N, H, H, stride, off,
                                                Ho, Ho, relu);
    else
      pw_conv_wmma_k<4><<<g2, 128, 0, stream>>>(X, Wp, Y, Cin, CinTot, ciBase,
                                                CoutTot, coBase, N, H, H, stride, off,
                                                Ho, Ho, relu);
  };

  auto axl = [&](float* O, int oT, int oB, const float* T2, int iT, int iB,
                 const float* wA, const float* wB, int prim, int C, int S) {
    long tot = (long)N * C * S;
    axpy_mixed_k<<<NB(tot), 256, 0, stream>>>(O, T2, wA, wB, prim, C, N, S, oT, oB, iT, iB);
  };

  auto dwl = [&](const float* X, int CinTot, int ciBase, int H,
                 const float* Wp, float* Y, int C, int Ho,
                 int k, int stride, int pad, int dil) {
    long tot = (long)N * C * Ho * Ho;
    dw_conv_k<<<NB(tot), 256, 0, stream>>>(X, Wp, Y, C, CinTot, ciBase, N, H, H,
                                           Ho, Ho, k, stride, pad, dil, /*relu=*/1);
  };

  auto sepl = [&](const St& h, const float* dw1, const float* pw1,
                  const float* dw2, const float* pw2, int k, int stride, int outH,
                  int C, float* O, int oT, int oB,
                  const float* wA, const float* wB, int prim) {
    int pad = k / 2, S = outH * outH;
    dwl(h.p, h.Ctot, h.cBase, h.H, dw1, u1, C, outH, k, stride, pad, 1);
    pwl(u1, C, C, 0, outH, pw1, C, C, 0, u2, 1, 0, /*relu=*/0);
    bnl(u2, C, S);
    dwl(u2, C, 0, outH, dw2, u1, C, outH, k, 1, pad, 1);   // relu folded into dw load
    pwl(u1, C, C, 0, outH, pw2, C, C, 0, t, 1, 0, 0);
    bnl(t, C, S);
    axl(O, oT, oB, t, C, 0, wA, wB, prim, C, S);
  };

  auto dill = [&](const St& h, const float* dww, const float* pww, int k, int stride,
                  int outH, int C, float* O, int oT, int oB,
                  const float* wA, const float* wB, int prim) {
    int pad = (k / 2) * 2, S = outH * outH;
    dwl(h.p, h.Ctot, h.cBase, h.H, dww, u1, C, outH, k, stride, pad, 2);
    pwl(u1, C, C, 0, outH, pww, C, C, 0, t, 1, 0, 0);
    bnl(t, C, S);
    axl(O, oT, oB, t, C, 0, wA, wB, prim, C, S);
  };

  // ---- stem ----
  {
    long tot = (long)N * 48 * 32 * 32;
    stem_conv_k<<<NB(tot), 256, 0, stream>>>(data, stem_w, B0, N, 32, 32);
    bnl(B0, 48, 32 * 32, stem_g, stem_b);
  }
  const float* s0 = B0; int s0C = 48, s0H = 32;
  const float* s1 = B0; int s1C = 48, s1H = 32;
  float* bufs[3] = { B0, B1, B2 };

  // ---- cell configs (replicates reference bookkeeping) ----
  struct Cfg { int cpp, cp, C; bool red, red_prev; };
  Cfg cfgs[8];
  {
    int cpp = 48, cp = 48, Ccur = 16; bool rp = false;
    for (int i = 0; i < 8; ++i) {
      bool red = (i == 2 || i == 5);
      if (red) Ccur *= 2;
      cfgs[i] = { cpp, cp, Ccur, red, rp };
      rp = red;
      cpp = cp; cp = 3 * Ccur;
    }
  }

  // ---- cells ----
  for (int li = 0; li < 8; ++li) {
    const Cfg cfg = cfgs[li];
    const int C = cfg.C;
    const bool red = cfg.red, red_prev = cfg.red_prev;
    const int inH = s1H;
    const int outH = red ? inH / 2 : inH;
    const int outS = outH * outH;

    // fetch params in leaf order
    const float *pre0a = nullptr, *pre0b = nullptr, *pre1w = nullptr;
    if (red_prev) { pre0a = nxt(); pre0b = nxt(); }
    else          { pre0a = nxt(); }
    pre1w = nxt();

    OpParams opp[9];
    {
      int oc = 0;
      for (int i = 0; i < 3; ++i)
        for (int j = 0; j < 2 + i; ++j) {
          int stride = (red && j < 2) ? 2 : 1;
          OpParams& q = opp[oc++];
          q.s3dw1 = nxt(); q.s3pw1 = nxt(); q.s3dw2 = nxt(); q.s3pw2 = nxt();
          q.d3dw  = nxt(); q.d3pw  = nxt();
          q.s5dw1 = nxt(); q.s5pw1 = nxt(); q.s5dw2 = nxt(); q.s5pw2 = nxt();
          q.d5dw  = nxt(); q.d5pw  = nxt();
          if (stride == 2) { q.skc1 = nxt(); q.skc2 = nxt(); }
          else             { q.skc1 = q.skc2 = nullptr; }
        }
    }

    // preprocess s0 -> s0p, s1 -> s1p (C channels)
    int s0pH;
    if (red_prev) {
      s0pH = s0H / 2;
      pwl(s0, s0C, s0C, 0, s0H, pre0a, C / 2, C, 0,     s0p, 2, 0, 1);
      pwl(s0, s0C, s0C, 0, s0H, pre0b, C / 2, C, C / 2, s0p, 2, 1, 1);
      bnl(s0p, C, s0pH * s0pH);
    } else {
      s0pH = s0H;
      pwl(s0, s0C, s0C, 0, s0H, pre0a, C, C, 0, s0p, 1, 0, 1);
      bnl(s0p, C, s0pH * s0pH);
    }
    pwl(s1, s1C, s1C, 0, inH, pre1w, C, C, 0, s1p, 1, 0, 1);
    bnl(s1p, C, inH * inH);

    // cell output buffer (3C channels @ outH), states written into slices
    float* OUT = nullptr;
    for (int bi = 0; bi < 3; ++bi)
      if (bufs[bi] != s0 && bufs[bi] != s1) { OUT = bufs[bi]; break; }
    (void)hipMemsetAsync(OUT, 0, sizeof(float) * (size_t)N * 3 * C * outS, stream);

    St sts[5];
    sts[0] = { s0p, C, 0, s0pH };
    sts[1] = { s1p, C, 0, inH  };
    int nst = 2, oc2 = 0;

    for (int i = 0; i < 3; ++i) {
      int obase = i * C;
      for (int j = 0; j < nst; ++j) {
        const St h = sts[j];
        int stride = (red && j < 2) ? 2 : 1;
        const OpParams& q = opp[oc2++];
        const float* wA = aw + (((long)li * 3 + i) * 2 + 0) * 32 + j * 8;
        const float* wB = wA + 32;
        const int S = outS;
        long tot = (long)N * C * S;

        // prim 1: max_pool_3x3
        pool3_k<<<NB(tot), 256, 0, stream>>>(h.p, t, C, h.Ctot, h.cBase, N,
                                             h.H, h.H, outH, outH, stride, 1);
        axl(OUT, 3 * C, obase, t, C, 0, wA, wB, 1, C, S);
        // prim 2: avg_pool_3x3
        pool3_k<<<NB(tot), 256, 0, stream>>>(h.p, t, C, h.Ctot, h.cBase, N,
                                             h.H, h.H, outH, outH, stride, 0);
        axl(OUT, 3 * C, obase, t, C, 0, wA, wB, 2, C, S);
        // prim 3: skip_connect
        if (stride == 1) {
          axl(OUT, 3 * C, obase, h.p, h.Ctot, h.cBase, wA, wB, 3, C, S);
        } else {
          pwl(h.p, C, h.Ctot, h.cBase, h.H, q.skc1, C / 2, C, 0,     t, 2, 0, 1);
          pwl(h.p, C, h.Ctot, h.cBase, h.H, q.skc2, C / 2, C, C / 2, t, 2, 1, 1);
          bnl(t, C, S);
          axl(OUT, 3 * C, obase, t, C, 0, wA, wB, 3, C, S);
        }
        // prim 4: sep_conv_3x3 ; prim 5: sep_conv_5x5
        sepl(h, q.s3dw1, q.s3pw1, q.s3dw2, q.s3pw2, 3, stride, outH, C,
             OUT, 3 * C, obase, wA, wB, 4);
        sepl(h, q.s5dw1, q.s5pw1, q.s5dw2, q.s5pw2, 5, stride, outH, C,
             OUT, 3 * C, obase, wA, wB, 5);
        // prim 6: dil_conv_3x3 ; prim 7: dil_conv_5x5
        dill(h, q.d3dw, q.d3pw, 3, stride, outH, C, OUT, 3 * C, obase, wA, wB, 6);
        dill(h, q.d5dw, q.d5pw, 5, stride, outH, C, OUT, 3 * C, obase, wA, wB, 7);
      }
      sts[nst++] = { OUT, 3 * C, i * C, outH };
    }

    // rotate states
    s0 = s1; s0C = s1C; s0H = s1H;
    s1 = OUT; s1C = 3 * C; s1H = outH;
  }

  // ---- head ----
  const float* fc_w = nxt();
  const float* fc_b = nxt();
  head_fc_k<<<NB((long)N * 10), 256, 0, stream>>>(s1, fc_w, fc_b, (float*)d_out,
                                                  N, s1C, s1H * s1H);
}
